// Attention_59811714564198
// MI455X (gfx1250) — matrix operand order repaired and checked by reading, not compiled, not run on hardware
//
#include <hip/hip_runtime.h>

// ---------------- constants (match reference) ----------------
#define BB 2
#define SS 2048
#define DD 1024
#define HH 16
#define DHH 64

typedef __attribute__((ext_vector_type(16))) __bf16 v16bf;
typedef __attribute__((ext_vector_type(8)))  __bf16 v8bf;
typedef __attribute__((ext_vector_type(8)))  float  v8f;
typedef __attribute__((ext_vector_type(4)))  float  v4f;

union V16 {
    v16bf v;
    v8bf  h[2];
};

__device__ __forceinline__ v8f wmma_bf16(v16bf a, v16bf b, v8f c) {
    return __builtin_amdgcn_wmma_f32_16x16x32_bf16(
        /*neg_a=*/false, a, /*neg_b=*/false, b,
        /*c_mod=*/(short)0, c, /*reuse_a=*/false, /*reuse_b=*/false);
}

// ---------------- weight cast + transpose: w[K][N] f32 -> wT[N][K] bf16 ----
__global__ void cast_transpose_w(const float* __restrict__ w,
                                 __bf16* __restrict__ wT, int K, int N) {
    int idx = blockIdx.x * blockDim.x + threadIdx.x;
    if (idx < K * N) {
        int k = idx / N;
        int n = idx - k * N;
        wT[(size_t)n * K + k] = (__bf16)w[idx];
    }
}

// ---------------- WMMA GEMM: C[M,N] = (A[M,K] @ WT[N,K]^T + bias) * oscale -
// Wave tile: 32 rows x 64 cols (2x4 register blocking, 8 WMMA per K-step).
// AMODE: 0 = A is fp32 row-major (convert in-register), 1 = A is bf16 row-major
// SMODE: 0 = store bf16 per-head [B,H,S,DH] (Q/K)
//        1 = store bf16 per-head transposed [B,H,DH,S] (V)
//        2 = store fp32 row-major [M,N] (final output)
template <int AMODE, int SMODE>
__global__ __launch_bounds__(256, 1) void gemm16_wmma(
    const void* __restrict__ A_, const __bf16* __restrict__ WT,
    const float* __restrict__ bias, void* __restrict__ out,
    int M, int N, int K, float oscale) {
    const int wave  = blockIdx.x * (blockDim.x >> 5) + (threadIdx.x >> 5);
    const int lane  = threadIdx.x & 31;
    const int lmod  = lane & 15;
    const int khalf = lane >> 4;               // 0 or 1
    const int tilesN = N >> 6;                 // 64-wide col tiles
    const int tm = wave / tilesN;
    const int tn = wave - tm * tilesN;
    if (tm * 32 >= M) return;                  // wave-uniform

    const int arow0 = tm * 32 + lmod;          // A row for this lane (subtile 0)
    const int col0  = tn * 64 + lmod;          // C column for this lane (subtile 0)

    v8f acc[2][4] = {};
    for (int k0 = 0; k0 < K; k0 += 32) {
        const int abase = k0 + (khalf ? 16 : 0);
        V16 a[2];
#pragma unroll
        for (int mi = 0; mi < 2; ++mi) {
            if (AMODE == 0) {
                const float* Ap =
                    (const float*)A_ + (size_t)(arow0 + mi * 16) * K + abase;
                v8bf h0, h1;
#pragma unroll
                for (int i = 0; i < 8; ++i) {
                    h0[i] = (__bf16)Ap[i];
                    h1[i] = (__bf16)Ap[8 + i];
                }
                a[mi].h[0] = h0;
                a[mi].h[1] = h1;
            } else {
                const __bf16* Ap =
                    (const __bf16*)A_ + (size_t)(arow0 + mi * 16) * K + abase;
                a[mi].h[0] = *(const v8bf*)(Ap);
                a[mi].h[1] = *(const v8bf*)(Ap + 8);
            }
        }
        v16bf b[4];
#pragma unroll
        for (int ni = 0; ni < 4; ++ni)
            b[ni] = *(const v16bf*)(WT + (size_t)(col0 + ni * 16) * K + k0 +
                                    (khalf ? 16 : 0));
#pragma unroll
        for (int mi = 0; mi < 2; ++mi)
#pragma unroll
            for (int ni = 0; ni < 4; ++ni)
                acc[mi][ni] = wmma_bf16(a[mi].v, b[ni], acc[mi][ni]);
    }

#pragma unroll
    for (int ni = 0; ni < 4; ++ni) {
        const int col = col0 + ni * 16;
        const float bv = bias[col];
#pragma unroll
        for (int mi = 0; mi < 2; ++mi) {
#pragma unroll
            for (int r = 0; r < 8; ++r) {
                const int rr = tm * 32 + mi * 16 + r + (khalf ? 8 : 0);
                const float val = (acc[mi][ni][r] + bv) * oscale;
                if (SMODE == 2) {
                    ((float*)out)[(size_t)rr * N + col] = val;
                } else {
                    const int b_ = rr / SS;
                    const int s_ = rr - b_ * SS;
                    const int h_ = col >> 6;   // col / DH
                    const int dh = col & 63;
                    if (SMODE == 0) {
                        ((__bf16*)out)[(((size_t)(b_ * HH + h_)) * SS + s_) * DHH +
                                       dh] = (__bf16)val;
                    } else {
                        ((__bf16*)out)[(((size_t)(b_ * HH + h_)) * DHH + dh) * SS +
                                       s_] = (__bf16)val;
                    }
                }
            }
        }
    }
}

// ---------------- flash attention (two 16-query tiles per wave) -----------
// Qh (pre-scaled by 1/sqrt(DH)), Kh: bf16 [B,H,S,DH]; VhT: bf16 [B,H,DH,S];
// attOut: bf16 [B,S,D]. Softmax runs in the transposed (A-layout) domain via
// a per-wave LDS round-trip; exp'd P lands directly in A-fragment order.
// Bulk key chunks run unmasked; only the single chunk straddling valid_len
// pays the pad-mask cndmasks.
__global__ __launch_bounds__(256, 1) void flash_attn(
    const __bf16* __restrict__ Qh, const __bf16* __restrict__ Kh,
    const __bf16* __restrict__ VhT, const int* __restrict__ valid_lens,
    __bf16* __restrict__ attOut) {
    const int wave  = blockIdx.x * (blockDim.x >> 5) + (threadIdx.x >> 5);
    const int lane  = threadIdx.x & 31;
    const int lmod  = lane & 15;
    const int khalf = lane >> 4;

    const int qblk = wave & 63;                 // 64 blocks of 32 queries
    const int bh   = wave >> 6;                 // b*H + h, 0..31
    const int b_   = bh / HH;
    const int h_   = bh - b_ * HH;
    const int vl   = valid_lens[b_];

    const __bf16* Q  = Qh  + (size_t)bh * SS * DHH;
    const __bf16* Kp = Kh  + (size_t)bh * SS * DHH;
    const __bf16* Vt = VhT + (size_t)bh * DHH * SS;

    // Q A-fragments: 2 q-tiles x 2 dh-chunks, held for the whole loop.
    V16 aq[2][2];
#pragma unroll
    for (int qt = 0; qt < 2; ++qt) {
        const __bf16* qp =
            Q + (size_t)(qblk * 32 + qt * 16 + lmod) * DHH + (khalf ? 16 : 0);
        aq[qt][0].h[0] = *(const v8bf*)(qp);
        aq[qt][0].h[1] = *(const v8bf*)(qp + 8);
        aq[qt][1].h[0] = *(const v8bf*)(qp + 32);
        aq[qt][1].h[1] = *(const v8bf*)(qp + 40);
    }

    // online softmax state: this lane owns row (qblk*32 + qt*16 + lmod)
    float m_row[2] = {-3.0e38f, -3.0e38f};
    float l_row[2] = {0.0f, 0.0f};
    v8f o[2][4] = {};

    // per-wave, per-qtile LDS score tile: 16 queries x 32 keys f32
    __shared__ float Sf[8][2][16 * 32];
    const int wid = (threadIdx.x >> 5) & 7;

    auto chunk = [&](int kb, bool masked) {
        // ---- K B-fragments for keys [kb,kb+16) and [kb+16,kb+32)
        const __bf16* kp0 = Kp + (size_t)(kb + lmod) * DHH + (khalf ? 16 : 0);
        const __bf16* kp1 = Kp + (size_t)(kb + 16 + lmod) * DHH + (khalf ? 16 : 0);
        const v16bf bk00 = *(const v16bf*)(kp0);
        const v16bf bk01 = *(const v16bf*)(kp0 + 32);
        const v16bf bk10 = *(const v16bf*)(kp1);
        const v16bf bk11 = *(const v16bf*)(kp1 + 32);

        const bool ok0 = (kb + lmod) < vl;
        const bool ok1 = (kb + 16 + lmod) < vl;

        // ---- both q-tiles' score WMMAs back-to-back so K frags die early
        v8f s[2][2];
#pragma unroll
        for (int qt = 0; qt < 2; ++qt) {
            v8f s0 = {}, s1 = {};
            s0 = wmma_bf16(aq[qt][0].v, bk00, s0);
            s0 = wmma_bf16(aq[qt][1].v, bk01, s0);
            s1 = wmma_bf16(aq[qt][0].v, bk10, s1);
            s1 = wmma_bf16(aq[qt][1].v, bk11, s1);
            s[qt][0] = s0;
            s[qt][1] = s1;
        }

        V16 ap[2];
#pragma unroll
        for (int qt = 0; qt < 2; ++qt) {
            // ---- (masked tail only) pad-mask + transpose store to LDS
            float* SP = &Sf[wid][qt][0];
#pragma unroll
            for (int r = 0; r < 8; ++r) {
                const int rr = r + (khalf ? 8 : 0);
                SP[rr * 32 + lmod]      = (!masked || ok0) ? s[qt][0][r] : -1.0e6f;
                SP[rr * 32 + 16 + lmod] = (!masked || ok1) ? s[qt][1][r] : -1.0e6f;
            }

            // ---- read transposed: this lane = row lmod,
            // keys {base..base+15}, base = khalf*16
            const float* xp = SP + lmod * 32 + (khalf ? 16 : 0);
            const v4f x0 = *(const v4f*)(xp);
            const v4f x1 = *(const v4f*)(xp + 4);
            const v4f x2 = *(const v4f*)(xp + 8);
            const v4f x3 = *(const v4f*)(xp + 12);

            // ---- in-lane row max, one cross-half combine
            float t = fmaxf(fmaxf(x0[0], x0[1]), fmaxf(x0[2], x0[3]));
            t = fmaxf(t, fmaxf(fmaxf(x1[0], x1[1]), fmaxf(x1[2], x1[3])));
            t = fmaxf(t, fmaxf(fmaxf(x2[0], x2[1]), fmaxf(x2[2], x2[3])));
            t = fmaxf(t, fmaxf(fmaxf(x3[0], x3[1]), fmaxf(x3[2], x3[3])));
            t = fmaxf(t, __shfl_xor(t, 16, 32));
            const float mn = fmaxf(m_row[qt], t);

            // ---- exp + sum; P lands directly in A-fragment element order
            v8bf lo, hi;
            float rs = 0.0f;
#pragma unroll
            for (int i = 0; i < 4; ++i) {
                const float p0 = __expf(x0[i] - mn); rs += p0; lo[i]     = (__bf16)p0;
                const float p1 = __expf(x1[i] - mn); rs += p1; lo[4 + i] = (__bf16)p1;
                const float p2 = __expf(x2[i] - mn); rs += p2; hi[i]     = (__bf16)p2;
                const float p3 = __expf(x3[i] - mn); rs += p3; hi[4 + i] = (__bf16)p3;
            }
            ap[qt].h[0] = lo;
            ap[qt].h[1] = hi;
            rs += __shfl_xor(rs, 16, 32);
            const float alpha = __expf(m_row[qt] - mn);
            l_row[qt] = l_row[qt] * alpha + rs;
            m_row[qt] = mn;

            // ---- broadcast alpha (row -> C-layout rows) and rescale O
#pragma unroll
            for (int r = 0; r < 8; ++r) {
                const float ar = __shfl(alpha, r + (khalf ? 8 : 0), 32);
                o[qt][0][r] *= ar; o[qt][1][r] *= ar;
                o[qt][2][r] *= ar; o[qt][3][r] *= ar;
            }
        }

        // ---- PV: 4 dh tiles, V B-frags shared by both q-tiles
        const size_t vrow = (size_t)lmod * SS + kb + (khalf ? 16 : 0);
        const v16bf vb0 = *(const v16bf*)(Vt + vrow);
        const v16bf vb1 = *(const v16bf*)(Vt + 16 * SS + vrow);
        const v16bf vb2 = *(const v16bf*)(Vt + 32 * SS + vrow);
        const v16bf vb3 = *(const v16bf*)(Vt + 48 * SS + vrow);
#pragma unroll
        for (int qt = 0; qt < 2; ++qt) {
            o[qt][0] = wmma_bf16(ap[qt].v, vb0, o[qt][0]);
            o[qt][1] = wmma_bf16(ap[qt].v, vb1, o[qt][1]);
            o[qt][2] = wmma_bf16(ap[qt].v, vb2, o[qt][2]);
            o[qt][3] = wmma_bf16(ap[qt].v, vb3, o[qt][3]);
        }
    };

    // bulk chunks (fully valid keys, no masking) + one masked tail chunk.
    const int kfull = vl & ~31;
    for (int kb = 0; kb < kfull; kb += 32) chunk(kb, false);
    if (vl > kfull) chunk(kfull, true);

    // ---- normalize (broadcast 1/l to C-layout rows) and store attOut
#pragma unroll
    for (int qt = 0; qt < 2; ++qt) {
        const float linv = 1.0f / l_row[qt];
#pragma unroll
        for (int r = 0; r < 8; ++r) {
            const float lr = __shfl(linv, r + (khalf ? 8 : 0), 32);
            const int s_ = qblk * 32 + qt * 16 + r + (khalf ? 8 : 0);
            const size_t base = ((size_t)b_ * SS + s_) * DD + h_ * DHH;
            attOut[base + 0 * 16 + lmod] = (__bf16)(o[qt][0][r] * lr);
            attOut[base + 1 * 16 + lmod] = (__bf16)(o[qt][1][r] * lr);
            attOut[base + 2 * 16 + lmod] = (__bf16)(o[qt][2][r] * lr);
            attOut[base + 3 * 16 + lmod] = (__bf16)(o[qt][3][r] * lr);
        }
    }
}

// ---------------- host launch ---------------------------------------------
extern "C" void kernel_launch(void* const* d_in, const int* in_sizes, int n_in,
                              void* d_out, int out_size, void* d_ws, size_t ws_size,
                              hipStream_t stream) {
    const float* xq = (const float*)d_in[0];
    const float* xk = (const float*)d_in[1];
    const float* xv = (const float*)d_in[2];
    const float* wq = (const float*)d_in[3];
    const float* bq = (const float*)d_in[4];
    const float* wk = (const float*)d_in[5];
    const float* bk = (const float*)d_in[6];
    const float* wv = (const float*)d_in[7];
    const float* bv = (const float*)d_in[8];
    const float* wo = (const float*)d_in[9];
    const float* bo = (const float*)d_in[10];
    const int* valid_lens = (const int*)d_in[11];

    char* ws = (char*)d_ws;
    const size_t MB = 1ull << 20;
    __bf16* wqT    = (__bf16*)(ws + 0 * MB);   // 1024*1024*2 = 2 MB each
    __bf16* wkT    = (__bf16*)(ws + 2 * MB);
    __bf16* wvT    = (__bf16*)(ws + 4 * MB);
    __bf16* woT    = (__bf16*)(ws + 6 * MB);
    __bf16* Qh     = (__bf16*)(ws + 8 * MB);   // [B,H,S,DH] bf16 = 8 MB
    __bf16* Kh     = (__bf16*)(ws + 16 * MB);  // 8 MB
    __bf16* VhT    = (__bf16*)(ws + 24 * MB);  // [B,H,DH,S] bf16 = 8 MB
    __bf16* attOut = (__bf16*)(ws + 32 * MB);  // [B,S,D] bf16 = 8 MB

    const int M = BB * SS;       // 4096
    const int N = DD;            // 1024
    const int K = DD;            // 1024

    // weight cast+transpose: 1M elements each
    const int cblk = (DD * DD + 255) / 256;
    cast_transpose_w<<<cblk, 256, 0, stream>>>(wq, wqT, K, N);
    cast_transpose_w<<<cblk, 256, 0, stream>>>(wk, wkT, K, N);
    cast_transpose_w<<<cblk, 256, 0, stream>>>(wv, wvT, K, N);
    cast_transpose_w<<<cblk, 256, 0, stream>>>(wo, woT, K, N);

    // projections: (M/32)*(N/64) = 2048 waves, 8 waves/block -> 256 blocks
    const int gblk = (M / 32) * (N / 64) / 8;
    const float qscale = 0.125f;   // 1/sqrt(DH), folded into Q
    gemm16_wmma<0, 0><<<gblk, 256, 0, stream>>>(xq, wqT, bq, Qh,  M, N, K, qscale);
    gemm16_wmma<0, 0><<<gblk, 256, 0, stream>>>(xk, wkT, bk, Kh,  M, N, K, 1.0f);
    gemm16_wmma<0, 1><<<gblk, 256, 0, stream>>>(xv, wvT, bv, VhT, M, N, K, 1.0f);

    // attention: B*H*(S/32) = 2048 waves -> 256 blocks
    flash_attn<<<256, 256, 0, stream>>>(Qh, Kh, VhT, valid_lens, attOut);

    // output projection to fp32 d_out
    gemm16_wmma<1, 2><<<gblk, 256, 0, stream>>>(attOut, woT, bo, d_out, M, N, K,
                                                1.0f);
}
